// MultiHeadAttention_67422396612828
// MI455X (gfx1250) — matrix operand order repaired and checked by reading, not compiled, not run on hardware
//
#include <hip/hip_runtime.h>

typedef __attribute__((ext_vector_type(16))) __bf16 bf16x16;
typedef __attribute__((ext_vector_type(8)))  __bf16 bf16x8;
typedef __attribute__((ext_vector_type(8)))  float  f32x8;
typedef __attribute__((ext_vector_type(4)))  unsigned int u32x4;
typedef __attribute__((ext_vector_type(8)))  int i32x8;
typedef __attribute__((ext_vector_type(4)))  int i32x4;

static constexpr int D_MODEL = 1024;
static constexpr int SEQ     = 2048;
static constexpr int HEADS   = 16;
static constexpr int DH      = 64;
static constexpr int BATCH   = 2;
static constexpr int M_TOT   = BATCH * SEQ;   // 4096

#if defined(__has_builtin)
#  if __has_builtin(__builtin_amdgcn_tensor_load_to_lds)
#    define HAVE_TDM 1
#  endif
#endif
#ifndef HAVE_TDM
#  define HAVE_TDM 0
#endif

// ---------------- WMMA helpers ----------------

__device__ __forceinline__ f32x8 wmma_bf16(bf16x16 a, bf16x16 b, f32x8 c) {
  // D = A(16x32 bf16) * B(32x16 bf16) + C(16x16 f32)
  return __builtin_amdgcn_wmma_f32_16x16x32_bf16(false, a, false, b, (short)0, c,
                                                 false, false);
}

__device__ __forceinline__ f32x8 zero8() {
  f32x8 z = {0.f, 0.f, 0.f, 0.f, 0.f, 0.f, 0.f, 0.f};
  return z;
}

// A fragment, 16x32, row-major source with leading dim `ld` (elements).
__device__ __forceinline__ bf16x16 frag_a(const __bf16* tile, int ld, int lane) {
  const int lr = lane & 15, half = lane >> 4;
  const __bf16* p = tile + lr * ld + half * 8;
  bf16x8 lo = *(const bf16x8*)p;
  bf16x8 hi = *(const bf16x8*)(p + 16);
  bf16x16 r;
#pragma unroll
  for (int i = 0; i < 8; ++i) { r[i] = lo[i]; r[i + 8] = hi[i]; }
  return r;
}

// B fragment, 32x16 (KxN), loaded from N-major storage (rows = N).
__device__ __forceinline__ bf16x16 frag_b(const __bf16* tile, int ld, int lane) {
  const int lr = lane & 15, half = lane >> 4;
  const __bf16* p = tile + lr * ld + half * 8;
  bf16x8 lo = *(const bf16x8*)p;
  bf16x8 hi = *(const bf16x8*)(p + 16);
  bf16x16 r;
#pragma unroll
  for (int i = 0; i < 8; ++i) { r[i] = lo[i]; r[i + 8] = hi[i]; }
  return r;
}

// stage 8 contiguous elements into LDS as bf16 (with f32->bf16 cvt if needed)
__device__ __forceinline__ void stage8(const float* src, __bf16* dst) {
  float4 x = *(const float4*)src;
  float4 y = *(const float4*)(src + 4);
  dst[0] = (__bf16)x.x; dst[1] = (__bf16)x.y; dst[2] = (__bf16)x.z; dst[3] = (__bf16)x.w;
  dst[4] = (__bf16)y.x; dst[5] = (__bf16)y.y; dst[6] = (__bf16)y.z; dst[7] = (__bf16)y.w;
}
__device__ __forceinline__ void stage8(const __bf16* src, __bf16* dst) {
  *(bf16x8*)dst = *(const bf16x8*)src;
}

// CDNA5 async global->LDS copy, 16 bytes per lane (ASYNCcnt path).
// LDS byte address = low 32 bits of the generic pointer (LDS aperture).
__device__ __forceinline__ void async_copy16(const void* g, void* lds_ptr) {
  const unsigned lds = (unsigned)(unsigned long long)lds_ptr;
  asm volatile("global_load_async_to_lds_b128 %0, %1, off"
               :: "v"(lds), "v"(g) : "memory");
}
__device__ __forceinline__ void async_wait0() {
  asm volatile("s_wait_asynccnt 0x0" ::: "memory");
}

#if HAVE_TDM
// Tensor Data Mover: DMA a 2D tile (16-bit elements) from global to LDS.
// Dims/strides in elements. D# layout per CDNA5 ISA 8.3/8.4.
__device__ __forceinline__ void tdm_load_2d(const void* gsrc, void* lds_dst,
                                            unsigned tile_d0, unsigned tile_d1,
                                            unsigned tensor_d0, unsigned tensor_d1,
                                            unsigned stride_d0) {
  const unsigned long long ga = (unsigned long long)gsrc;
  u32x4 g0;
  g0[0] = 1u;                                       // count=1, user mode
  g0[1] = (unsigned)(unsigned long long)lds_dst;    // lds_addr (bytes)
  g0[2] = (unsigned)(ga & 0xffffffffu);             // global_addr[31:0]
  g0[3] = (unsigned)((ga >> 32) & 0x1ffffffu) | (2u << 30);  // addr[56:32], type=2
  i32x8 g1;
  g1[0] = (int)(1u << 16);                          // data_size=1 (2 bytes)
  g1[1] = (int)((tensor_d0 & 0xffffu) << 16);                        // dim0[15:0]
  g1[2] = (int)(((tensor_d0 >> 16) & 0xffffu) | ((tensor_d1 & 0xffffu) << 16));
  g1[3] = (int)(((tensor_d1 >> 16) & 0xffffu) | ((tile_d0 & 0xffffu) << 16));
  g1[4] = (int)(tile_d1 & 0xffffu);                 // tile_dim1; tile_dim2=0
  g1[5] = (int)stride_d0;                           // dim0 stride[31:0]
  g1[6] = 0;
  g1[7] = 0;
  i32x4 z4 = {0, 0, 0, 0};
#if __clang_major__ >= 23
  i32x8 z8 = {0, 0, 0, 0, 0, 0, 0, 0};
  __builtin_amdgcn_tensor_load_to_lds(g0, g1, z4, z4, z8, 0);
#else
  __builtin_amdgcn_tensor_load_to_lds(g0, g1, z4, z4, 0);
#endif
}
#endif

// xor-reductions across the 16-lane half-wave (rows live on 16 lanes)
__device__ __forceinline__ float rmax16(float v) {
  v = fmaxf(v, __int_as_float(__builtin_amdgcn_ds_swizzle(__float_as_int(v), 0x041f)));
  v = fmaxf(v, __int_as_float(__builtin_amdgcn_ds_swizzle(__float_as_int(v), 0x081f)));
  v = fmaxf(v, __int_as_float(__builtin_amdgcn_ds_swizzle(__float_as_int(v), 0x101f)));
  v = fmaxf(v, __int_as_float(__builtin_amdgcn_ds_swizzle(__float_as_int(v), 0x201f)));
  return v;
}
__device__ __forceinline__ float rsum16(float v) {
  v += __int_as_float(__builtin_amdgcn_ds_swizzle(__float_as_int(v), 0x041f));
  v += __int_as_float(__builtin_amdgcn_ds_swizzle(__float_as_int(v), 0x081f));
  v += __int_as_float(__builtin_amdgcn_ds_swizzle(__float_as_int(v), 0x101f));
  v += __int_as_float(__builtin_amdgcn_ds_swizzle(__float_as_int(v), 0x201f));
  return v;
}

// ---------------- GEMM: C[M,N] = A[M,K] * W[N,K]^T + bias ----------------
// MODE 0: out bf16 [B,H,S,dh]; MODE 1: out bf16 [B,H,dh,S]; MODE 2: out f32 [M,N]
template <int MODE, typename AT>
__global__ void __launch_bounds__(256)
mha_gemm(const AT* __restrict__ A, const float* __restrict__ W,
         const float* __restrict__ bias, void* __restrict__ outv) {
  __shared__ __bf16 As[64 * 64];    // 8 KB
  __shared__ __bf16 Bs[128 * 64];   // 16 KB

  const int tid = threadIdx.x;
  const int lane = tid & 31, wid = tid >> 5;
  const int m0 = blockIdx.x * 64, n0 = blockIdx.y * 128;
  const int wm = (wid & 1) * 32, wn = (wid >> 1) * 32;

  f32x8 acc[2][2];
#pragma unroll
  for (int i = 0; i < 2; ++i)
#pragma unroll
    for (int j = 0; j < 2; ++j) acc[i][j] = zero8();

  const int ar = tid >> 2, ac = (tid & 3) * 16;   // 64 rows x 4 col-segs of 16

  for (int kk = 0; kk < D_MODEL; kk += 64) {
    const AT* asrc = A + (size_t)(m0 + ar) * D_MODEL + kk + ac;
    if (MODE == 2) {   // bf16 A: DMA straight into LDS, no VGPR round-trip
      async_copy16(asrc, &As[ar * 64 + ac]);
      async_copy16(asrc + 8, &As[ar * 64 + ac + 8]);
    } else {
      stage8(asrc, &As[ar * 64 + ac]);
      stage8(asrc + 8, &As[ar * 64 + ac + 8]);
    }
#pragma unroll
    for (int seg = 0; seg < 2; ++seg) {
      const int br = seg * 64 + ar;
      const float* wsrc = W + (size_t)(n0 + br) * D_MODEL + kk + ac;
      stage8(wsrc, &Bs[br * 64 + ac]);
      stage8(wsrc + 8, &Bs[br * 64 + ac + 8]);
    }
    if (kk + 64 < D_MODEL) {   // prefetch next K-step (global_prefetch_b8)
      __builtin_prefetch(A + (size_t)(m0 + ar) * D_MODEL + kk + 64 + ac, 0, 3);
      __builtin_prefetch(W + (size_t)(n0 + ar) * D_MODEL + kk + 64 + ac, 0, 3);
      __builtin_prefetch(W + (size_t)(n0 + 64 + ar) * D_MODEL + kk + 64 + ac, 0, 3);
    }
    if (MODE == 2) async_wait0();
    __syncthreads();

#pragma unroll
    for (int kb2 = 0; kb2 < 2; ++kb2) {
      const int k0 = kb2 * 32;
      bf16x16 a0 = frag_a(&As[wm * 64 + k0], 64, lane);
      bf16x16 a1 = frag_a(&As[(wm + 16) * 64 + k0], 64, lane);
      bf16x16 b0 = frag_b(&Bs[wn * 64 + k0], 64, lane);
      bf16x16 b1 = frag_b(&Bs[(wn + 16) * 64 + k0], 64, lane);
      acc[0][0] = wmma_bf16(a0, b0, acc[0][0]);
      acc[0][1] = wmma_bf16(a0, b1, acc[0][1]);
      acc[1][0] = wmma_bf16(a1, b0, acc[1][0]);
      acc[1][1] = wmma_bf16(a1, b1, acc[1][1]);
    }
    __syncthreads();
  }

  const int lr = lane & 15, half = lane >> 4;
#pragma unroll
  for (int mt = 0; mt < 2; ++mt) {
#pragma unroll
    for (int nt = 0; nt < 2; ++nt) {
      const int n = n0 + wn + nt * 16 + lr;
      const float bv = bias[n];
#pragma unroll
      for (int i = 0; i < 8; ++i) {
        const int m = m0 + wm + mt * 16 + half * 8 + i;
        const float v = acc[mt][nt][i] + bv;
        if (MODE == 2) {
          ((float*)outv)[(size_t)m * D_MODEL + n] = v;
        } else {
          const int b = m >> 11, s = m & (SEQ - 1);
          const int h = n >> 6, d = n & (DH - 1);
          size_t idx;
          if (MODE == 0)
            idx = ((size_t)(b * HEADS + h) * SEQ + s) * DH + d;
          else
            idx = ((size_t)(b * HEADS + h) * DH + d) * SEQ + s;
          ((__bf16*)outv)[idx] = (__bf16)v;
        }
      }
    }
  }
}

// ---------------- Flash attention ----------------
// Q,K: bf16 [B*H, S, 64]; Vt: bf16 [B*H, 64, S]; O: bf16 [B, S, H*64]
__global__ void __launch_bounds__(256)
mha_flash(const __bf16* __restrict__ Q, const __bf16* __restrict__ K,
          const __bf16* __restrict__ Vt, __bf16* __restrict__ O) {
  __shared__ __bf16 Ks[2][64 * 64];     // double-buffered, 2 x 8 KB
  __shared__ __bf16 Vs[2][64 * 64];     // double-buffered, 2 x 8 KB
  __shared__ __bf16 Ps[8 * 16 * 64];    // per-wave 16q x 64k, 16 KB

  const int tid = threadIdx.x, lane = tid & 31, wid = tid >> 5;
  const int lr = lane & 15, half = lane >> 4;
  const int bh = blockIdx.x;
  const int q0 = blockIdx.y * 128 + wid * 16;

  const __bf16* qbase = Q + ((size_t)bh * SEQ + q0) * DH;
  bf16x16 aq0 = frag_a(qbase, DH, lane);        // d = 0..31
  bf16x16 aq1 = frag_a(qbase + 32, DH, lane);   // d = 32..63

  f32x8 accO[4];
#pragma unroll
  for (int t = 0; t < 4; ++t) accO[t] = zero8();
  float mrow[8], lrow[8];
#pragma unroll
  for (int i = 0; i < 8; ++i) { mrow[i] = -1e30f; lrow[i] = 0.f; }

  __bf16* pw = &Ps[wid * 16 * 64];
  const int sr = tid >> 2, sc = (tid & 3) * 16;   // 64 rows x 4 chunks of 16

  // issue the staged copy for key-block `kb` into buffer `buf`
  auto stage = [&](int buf, int kb) {
#if HAVE_TDM
    if (wid == 0)
      tdm_load_2d(K + ((size_t)bh * SEQ + kb) * DH, &Ks[buf][0],
                  /*tile*/ DH, 64, /*tensor*/ DH, SEQ, /*stride*/ DH);
    else if (wid == 1)
      tdm_load_2d(Vt + (size_t)bh * DH * SEQ + kb, &Vs[buf][0],
                  /*tile*/ 64, DH, /*tensor*/ SEQ, DH, /*stride*/ SEQ);
#else
    async_copy16(K + ((size_t)bh * SEQ + kb + sr) * DH + sc, &Ks[buf][sr * 64 + sc]);
    async_copy16(Vt + ((size_t)bh * DH + sr) * SEQ + kb + sc, &Vs[buf][sr * 64 + sc]);
#endif
  };
  auto stage_wait = [&]() {
#if HAVE_TDM
    if (wid <= 1) __builtin_amdgcn_s_wait_tensorcnt(0);
#else
    async_wait0();
#endif
  };

  stage(0, 0);
  stage_wait();
  __syncthreads();

  for (int kb = 0; kb < SEQ; kb += 64) {
    const int cur = (kb >> 6) & 1;
    if (kb + 64 < SEQ) stage(cur ^ 1, kb + 64);   // overlap DMA with compute

    // scores: 16q x 64k, reduce over d=64 (two WMMAs per 16-key tile)
    f32x8 sc4[4];
#pragma unroll
    for (int kt = 0; kt < 4; ++kt) {
      sc4[kt] = zero8();
      sc4[kt] = wmma_bf16(aq0, frag_b(&Ks[cur][kt * 16 * 64], 64, lane), sc4[kt]);
      sc4[kt] = wmma_bf16(aq1, frag_b(&Ks[cur][kt * 16 * 64 + 32], 64, lane), sc4[kt]);
    }

    // online softmax per q-row (rows live across the 16-lane half)
#pragma unroll
    for (int i = 0; i < 8; ++i) {
      const float v0 = sc4[0][i] * 0.125f;   // 1/sqrt(64)
      const float v1 = sc4[1][i] * 0.125f;
      const float v2 = sc4[2][i] * 0.125f;
      const float v3 = sc4[3][i] * 0.125f;
      const float rmax = rmax16(fmaxf(fmaxf(v0, v1), fmaxf(v2, v3)));
      const float mnew = fmaxf(mrow[i], rmax);
      const float p0 = __expf(v0 - mnew);
      const float p1 = __expf(v1 - mnew);
      const float p2 = __expf(v2 - mnew);
      const float p3 = __expf(v3 - mnew);
      const float alpha = __expf(mrow[i] - mnew);
      lrow[i] = lrow[i] * alpha + rsum16(p0 + p1 + p2 + p3);
      mrow[i] = mnew;
#pragma unroll
      for (int t = 0; t < 4; ++t) accO[t][i] *= alpha;
      // re-layout C-tile -> A-fragment via per-wave LDS (LDS in-order per wave)
      const int row = (half * 8 + i) * 64;
      pw[row + lr] = (__bf16)p0;
      pw[row + 16 + lr] = (__bf16)p1;
      pw[row + 32 + lr] = (__bf16)p2;
      pw[row + 48 + lr] = (__bf16)p3;
    }

    bf16x16 ap0 = frag_a(pw, 64, lane);        // keys 0..31
    bf16x16 ap1 = frag_a(pw + 32, 64, lane);   // keys 32..63
#pragma unroll
    for (int dt = 0; dt < 4; ++dt) {
      accO[dt] = wmma_bf16(ap0, frag_b(&Vs[cur][dt * 16 * 64], 64, lane), accO[dt]);
      accO[dt] = wmma_bf16(ap1, frag_b(&Vs[cur][dt * 16 * 64 + 32], 64, lane), accO[dt]);
    }

    stage_wait();       // next buffer landed (overlapped with the WMMAs above)
    __syncthreads();
  }

  const int b = bh >> 4, h = bh & 15;
  float inv[8];
#pragma unroll
  for (int i = 0; i < 8; ++i) inv[i] = 1.f / lrow[i];
#pragma unroll
  for (int t = 0; t < 4; ++t) {
#pragma unroll
    for (int i = 0; i < 8; ++i) {
      const int s = q0 + half * 8 + i;
      O[((size_t)b * SEQ + s) * D_MODEL + h * DH + t * 16 + lr] =
          (__bf16)(accO[t][i] * inv[i]);
    }
  }
}

// ---------------- launch ----------------
extern "C" void kernel_launch(void* const* d_in, const int* in_sizes, int n_in,
                              void* d_out, int out_size, void* d_ws, size_t ws_size,
                              hipStream_t stream) {
  const float* q  = (const float*)d_in[0];
  const float* k  = (const float*)d_in[1];
  const float* v  = (const float*)d_in[2];
  const float* Wq = (const float*)d_in[3];
  const float* bq = (const float*)d_in[4];
  const float* Wk = (const float*)d_in[5];
  const float* bk = (const float*)d_in[6];
  const float* Wv = (const float*)d_in[7];
  const float* bv = (const float*)d_in[8];
  const float* Wo = (const float*)d_in[9];
  const float* bo = (const float*)d_in[10];

  char* ws = (char*)d_ws;
  const size_t SEG = (size_t)BATCH * HEADS * SEQ * DH * sizeof(__bf16);  // 8 MB
  __bf16* Qb = (__bf16*)(ws);
  __bf16* Kb = (__bf16*)(ws + SEG);
  __bf16* Vb = (__bf16*)(ws + 2 * SEG);
  __bf16* Ab = (__bf16*)(ws + 3 * SEG);

  dim3 gg(M_TOT / 64, D_MODEL / 128);
  mha_gemm<0, float><<<gg, 256, 0, stream>>>(q, Wq, bq, (void*)Qb);
  mha_gemm<0, float><<<gg, 256, 0, stream>>>(k, Wk, bk, (void*)Kb);
  mha_gemm<1, float><<<gg, 256, 0, stream>>>(v, Wv, bv, (void*)Vb);
  mha_flash<<<dim3(BATCH * HEADS, SEQ / 128), 256, 0, stream>>>(Qb, Kb, Vb, Ab);
  mha_gemm<2, __bf16><<<gg, 256, 0, stream>>>(Ab, Wo, bo, d_out);
}
